// Regressor1_41944650613179
// MI455X (gfx1250) — matrix operand order repaired and checked by reading, not compiled, not run on hardware
//
#include <hip/hip_runtime.h>

typedef __attribute__((ext_vector_type(16))) __bf16 v16bf;
typedef __attribute__((ext_vector_type(8)))  __bf16 v8bf;
typedef __attribute__((ext_vector_type(4)))  __bf16 v4bf;
typedef __attribute__((ext_vector_type(8)))  float  v8f;

#define Bn 2
#define Cn 128
#define Hn 448
#define Wn 608
#define Kn 128
#define SLOPE 0.01f
#define CHUNK 128
#define SWA 136   /* Wc hi/lo row stride (bf16) */
#define SWB 136   /* X^T  row stride (bf16)     */
#define SA  132   /* zbuf (fp32) row stride     */

__device__ __forceinline__ v16bf ld_frag(const __bf16* p0, const __bf16* p1) {
  v8bf a = *(const v8bf*)p0;   // 16B LDS load
  v8bf b = *(const v8bf*)p1;   // 16B LDS load
  return __builtin_shufflevector(a, b, 0,1,2,3,4,5,6,7,8,9,10,11,12,13,14,15);
}

__global__ __launch_bounds__(256) void regressor_fused_kernel(
    const float* __restrict__ x,  const float* __restrict__ x_gt,
    const float* __restrict__ Wc, const float* __restrict__ bc,
    const float* __restrict__ Wr, const float* __restrict__ br,
    const float* __restrict__ Wm, const float* __restrict__ bm,
    float* __restrict__ out)
{
  // [WcHi | WcLo | XsThi | XsTlo | bcs | red]; Wc region aliased by fp32 z-tile later.
  __shared__ __align__(16) unsigned char smem[4 * Kn * SWA * 2 + Kn * 4 + 3 * 256 * 4];
  __bf16* wchi = (__bf16*)smem;            // [Kn][SWA]  row-major (k, c)
  __bf16* wclo = wchi + Kn * SWA;
  __bf16* xshi = wclo + Kn * SWA;          // [CHUNK][SWB]  transposed (col, c)
  __bf16* xslo = xshi + CHUNK * SWB;
  float*  bcs  = (float*)(xslo + CHUNK * SWB);
  float*  red  = bcs + Kn;                 // [3][256] reduction scratch
  float*  zbuf = (float*)smem;             // [Kn][SA] fp32, aliases Wc region

  const int tid   = threadIdx.x;
  const int lane  = tid & 31;
  const int wave  = tid >> 5;              // 0..7 (wave32)
  const int l15   = lane & 15;
  const int hi    = lane >> 4;
  const int wbase = blockIdx.x * CHUNK;
  const int h     = blockIdx.y;
  const int b     = blockIdx.z;

  // ---------------- Phase 1a: Wc[h] -> bf16 hi/lo, row-major ----------------
  const float* wcrow = Wc + (size_t)h * Kn * Cn;
  for (int idx = tid; idx < Kn * 32; idx += 256) {
    const int k  = idx >> 5;
    const int c4 = idx & 31;
    const float4 wv = *(const float4*)(wcrow + k * Cn + 4 * c4);
    v4bf hi4, lo4;
    hi4.x = (__bf16)wv.x; lo4.x = (__bf16)(wv.x - (float)hi4.x);
    hi4.y = (__bf16)wv.y; lo4.y = (__bf16)(wv.y - (float)hi4.y);
    hi4.z = (__bf16)wv.z; lo4.z = (__bf16)(wv.z - (float)hi4.z);
    hi4.w = (__bf16)wv.w; lo4.w = (__bf16)(wv.w - (float)hi4.w);
    *(v4bf*)(wchi + k * SWA + 4 * c4) = hi4;
    *(v4bf*)(wclo + k * SWA + 4 * c4) = lo4;
  }

  // ---------------- Phase 1b: X tile -> bf16 hi/lo, TRANSPOSED [col][c] ----------------
  for (int idx = tid; idx < 32 * CHUNK; idx += 256) {
    const int g   = idx >> 7;              // 4-channel group: c = 4g..4g+3
    const int col = idx & 127;
    const int w   = wbase + col;
    float xv[4] = {0.f, 0.f, 0.f, 0.f};
    if (w < Wn) {
#pragma unroll
      for (int j = 0; j < 4; ++j)
        xv[j] = x[(((size_t)b * Cn + (4 * g + j)) * Hn + h) * Wn + w];
    }
    v4bf hi4, lo4;
    hi4.x = (__bf16)xv[0]; lo4.x = (__bf16)(xv[0] - (float)hi4.x);
    hi4.y = (__bf16)xv[1]; lo4.y = (__bf16)(xv[1] - (float)hi4.y);
    hi4.z = (__bf16)xv[2]; lo4.z = (__bf16)(xv[2] - (float)hi4.z);
    hi4.w = (__bf16)xv[3]; lo4.w = (__bf16)(xv[3] - (float)hi4.w);
    *(v4bf*)(xshi + col * SWB + 4 * g) = hi4;
    *(v4bf*)(xslo + col * SWB + 4 * g) = lo4;
  }
  if (tid < Kn) bcs[tid] = bc[h * Kn + tid];
  __syncthreads();

  // ---------------- Phase 2: cls = Wc @ X via bf16x3 V_WMMA_F32_16X16X32_BF16 ----------------
  // x*w ~= xh*wh + xh*wl + xl*wh ; fp32 accumulate. Wave owns classes [16*wave,16*wave+16).
  v8f acc[8];
#pragma unroll
  for (int nt = 0; nt < 8; ++nt) acc[nt] = (v8f){0.f,0.f,0.f,0.f,0.f,0.f,0.f,0.f};

  const int rowA = 16 * wave + l15;
#pragma unroll
  for (int s = 0; s < 4; ++s) {           // K slabs of 32 channels
    const int cs = 32 * s;
    // A frag (16x32 bf16): lanes 0-15 -> K=cs+0..7 & cs+16..23 ; lanes 16-31 -> +8
    const __bf16* pah = wchi + rowA * SWA + cs + 8 * hi;
    const __bf16* pal = wclo + rowA * SWA + cs + 8 * hi;
    const v16bf aHi = ld_frag(pah, pah + 16);
    const v16bf aLo = ld_frag(pal, pal + 16);
#pragma unroll
    for (int nt = 0; nt < 8; ++nt) {
      const int col = 16 * nt + l15;
      // B frag (32x16 bf16): lanes 0-15 -> K=cs+0..7 & cs+16..23 ; lanes 16-31 -> +8
      const __bf16* pbh = xshi + col * SWB + cs + 8 * hi;
      const __bf16* pbl = xslo + col * SWB + cs + 8 * hi;
      const v16bf bHi = ld_frag(pbh, pbh + 16);
      const v16bf bLo = ld_frag(pbl, pbl + 16);
      acc[nt] = __builtin_amdgcn_wmma_f32_16x16x32_bf16(false, aHi, false, bHi,
                                                        (short)0, acc[nt], false, false);
      acc[nt] = __builtin_amdgcn_wmma_f32_16x16x32_bf16(false, aHi, false, bLo,
                                                        (short)0, acc[nt], false, false);
      acc[nt] = __builtin_amdgcn_wmma_f32_16x16x32_bf16(false, aLo, false, bHi,
                                                        (short)0, acc[nt], false, false);
    }
  }
  __syncthreads();   // all Wc-region reads complete -> safe to alias as z-buffer

  // ---------------- Phase 3: bias + leaky_relu, spill z tile to LDS ----------------
#pragma unroll
  for (int nt = 0; nt < 8; ++nt) {
    const int col = 16 * nt + l15;
#pragma unroll
    for (int v = 0; v < 8; ++v) {
      const int r = 16 * wave + v + 8 * hi;   // C/D layout: VGPR v, halves M=v / M=v+8
      float z = acc[nt][v] + bcs[r];
      z = (z > 0.f) ? z : SLOPE * z;
      zbuf[r * SA + col] = z;
    }
  }
  __syncthreads();

  // ---------------- Phase 4: split-reduction softmax / loss / reg / mask ----------------
  // 2 threads per column: thread (col, half) reduces 64 classes / 64 channels; partials
  // combined via LDS. All 256 threads participate; stores done by half 0.
  {
    const int col  = tid & 127;
    const int half = tid >> 7;
    const int k0   = 64 * half;
    const int w    = wbase + col;
    const size_t pix = ((size_t)b * Hn + h) * Wn + (w < Wn ? w : (Wn - 1));

    const float gt = x_gt[pix];            // clamped addr; value unused for padded cols
    int ind = (int)(gt * (float)Kn);
    ind = ind < 0 ? 0 : (ind > Kn - 1 ? Kn - 1 : ind);
    const float* wr = Wr + ((size_t)h * Kn + ind) * Cn;
    __builtin_prefetch(wr + k0, 0, 1);     // global_prefetch_b8: hide the gather latency

    float zm = -3.402823466e38f;
    for (int k = 0; k < 64; ++k) zm = fmaxf(zm, zbuf[(k0 + k) * SA + col]);
    red[tid] = zm;                         // slot 0
    __syncthreads();
    const float zmax = fmaxf(red[col], red[col + 128]);

    float Sp = 0.f;
    for (int k = 0; k < 64; ++k) Sp += expf(zbuf[(k0 + k) * SA + col] - zmax);
    red[256 + tid] = Sp;                   // slot 1
    __syncthreads();
    const float S = red[256 + col] + red[256 + col + 128];
    const float maxp = 1.f / S;            // max_k softmax(z)_k == exp(0)/S

    float Tp = 0.f;                        // sum_k exp(p_k - max_p)
    for (int k = 0; k < 64; ++k) {
      const float p = expf(zbuf[(k0 + k) * SA + col] - zmax) * maxp;
      Tp += expf(p - maxp);
    }
    red[512 + tid] = Tp;                   // slot 2

    // reg (at ind) + mask: 64-channel partial dots, x reconstructed hi+lo.
    // Slot 0 rewrite is ordered after its last read by a sync above; slot 1's last read has no sync after it.
    float accR = 0.f, accM = 0.f;
    const __bf16* xh = xshi + col * SWB + k0;
    const __bf16* xl = xslo + col * SWB + k0;
    const float* wrh = wr + k0;
    const float* wmh = Wm + k0;
    for (int c = 0; c < 64; ++c) {
      const float xv = (float)xh[c] + (float)xl[c];
      accR = fmaf(xv, wrh[c], accR);
      accM = fmaf(xv, wmh[c], accM);
    }
    red[tid]       = accR;                 // slot 0 (reuse)
    red[256 + tid] = accM;                 // slot 1 (reuse)
    __syncthreads();

    if (half == 0 && w < Wn) {
      const float T     = red[512 + col] + red[512 + col + 128];
      const float p_ind = expf(zbuf[ind * SA + col] - zmax) * maxp;
      const float loss  = maxp + logf(T) - p_ind;   // -(p_ind - (maxp + log T))

      const float aR = red[col] + red[col + 128] + br[h * Kn + ind];
      float aM = red[256 + col] + red[256 + col + 128] + bm[0];
      aM = (aM > 0.f) ? aM : SLOPE * aM;
      const float x_out = ((float)ind + aR) * (1.f / (float)Kn);

      out[pix]                            = x_out;  // x_out [B,1,H,W]
      out[(size_t)Bn * Hn * Wn + pix]     = aM;     // mask  [B,1,H,W]
      out[2 * (size_t)Bn * Hn * Wn + pix] = loss;   // loss  [B,H,W]
    }
  }
}

extern "C" void kernel_launch(void* const* d_in, const int* in_sizes, int n_in,
                              void* d_out, int out_size, void* d_ws, size_t ws_size,
                              hipStream_t stream) {
  const float* x    = (const float*)d_in[0];
  const float* x_gt = (const float*)d_in[1];
  const float* Wc   = (const float*)d_in[2];
  const float* bc   = (const float*)d_in[3];
  const float* Wr   = (const float*)d_in[4];
  const float* br   = (const float*)d_in[5];
  const float* Wm   = (const float*)d_in[6];
  const float* bm   = (const float*)d_in[7];

  dim3 grid((Wn + CHUNK - 1) / CHUNK, Hn, Bn);   // (5, 448, 2)
  regressor_fused_kernel<<<grid, 256, 0, stream>>>(
      x, x_gt, Wc, bc, Wr, br, Wm, bm, (float*)d_out);
}
